// SpixConvDenoiser_58892591563422
// MI455X (gfx1250) — matrix operand order repaired and checked by reading, not compiled, not run on hardware
//
#include <hip/hip_runtime.h>
#include <hip/hip_bf16.h>

// Problem constants (match reference setup_inputs)
#define T_   4
#define C_   3
#define H_   192
#define W_   192
#define S_   300
#define SP_  320          // S padded to multiple of 32 for WMMA K
#define KS_  7
#define PAD_ 3
#define K2_  49
#define HW_  (H_*W_)
#define HP_  (H_ + 6)     // padded rows: 3 top + 3 bottom = 198
#define WP_  (W_ + 16)    // padded cols: 3 left + 13 right = 208 (covers w0+28)
#define SCALE_ 10.0f
#define EPS_   1e-5f

typedef _Float16 v16h __attribute__((ext_vector_type(16)));
typedef _Float16 v8h  __attribute__((ext_vector_type(8)));
typedef float    v8f  __attribute__((ext_vector_type(8)));

#define CAT16(lo, hi) __builtin_shufflevector(lo, hi, 0,1,2,3,4,5,6,7,8,9,10,11,12,13,14,15)

static __device__ inline v8f zero8f() {
    v8f z;
#pragma unroll
    for (int i = 0; i < 8; ++i) z[i] = 0.0f;
    return z;
}

// ---------------------------------------------------------------------------
// Kernel 0: b128 zero-fill of the padded sims buffer (border must read 0)
// ---------------------------------------------------------------------------
__global__ __launch_bounds__(256) void k_fill0(float4* __restrict__ p, size_t n16) {
    const float4 z = make_float4(0.f, 0.f, 0.f, 0.f);
    size_t stride = (size_t)gridDim.x * 256;
    for (size_t i = blockIdx.x * 256ull + threadIdx.x; i < n16; i += stride)
        p[i] = z;
}

// ---------------------------------------------------------------------------
// Kernel 1: zero the segment accumulators (T*S*4 floats: c0,c1,c2,count)
// ---------------------------------------------------------------------------
__global__ __launch_bounds__(256) void k_zero(float* __restrict__ sums, int n) {
    int i = blockIdx.x * 256 + threadIdx.x;
    if (i < n) sums[i] = 0.0f;
}

// ---------------------------------------------------------------------------
// Kernel 2: segment-sum accumulate (superpixel pooling) via float atomics
// ---------------------------------------------------------------------------
__global__ __launch_bounds__(256) void k_accum(const float* __restrict__ x,
                                               const int* __restrict__ spix,
                                               float* __restrict__ sums) {
    int pix = blockIdx.x * 256 + threadIdx.x;          // 0 .. T*HW-1 exact
    int t = pix / HW_;
    int hw = pix - t * HW_;
    int s = spix[pix];
    float f0 = x[(size_t)(t * C_ + 0) * HW_ + hw];
    float f1 = x[(size_t)(t * C_ + 1) * HW_ + hw];
    float f2 = x[(size_t)(t * C_ + 2) * HW_ + hw];
    float* base = sums + ((size_t)t * S_ + s) * 4;
    atomicAdd(base + 0, f0);
    atomicAdd(base + 1, f1);
    atomicAdd(base + 2, f2);
    atomicAdd(base + 3, 1.0f);
}

// ---------------------------------------------------------------------------
// Kernel 3: finalize means -> float4 (mu0, mu1, mu2, |mu|^2; absent -> 1e30)
// ---------------------------------------------------------------------------
__global__ __launch_bounds__(256) void k_means(const float* __restrict__ sums,
                                               float4* __restrict__ means) {
    int i = blockIdx.x * 256 + threadIdx.x;
    if (i >= T_ * S_) return;
    const float* s4 = sums + (size_t)i * 4;
    float cnt = s4[3];
    float4 mv;
    if (cnt > 0.0f) {
        float inv = 1.0f / cnt;
        mv.x = s4[0] * inv;
        mv.y = s4[1] * inv;
        mv.z = s4[2] * inv;
        mv.w = mv.x * mv.x + mv.y * mv.y + mv.z * mv.z;
    } else {
        mv.x = 0.0f; mv.y = 0.0f; mv.z = 0.0f; mv.w = 1e30f;
    }
    means[i] = mv;
}

// ---------------------------------------------------------------------------
// Kernel 4: per-pixel softmax over superpixels -> sims (f16, padded image)
// Block = 256 pixels of one frame; means cached in LDS.
// ---------------------------------------------------------------------------
__global__ __launch_bounds__(256) void k_sims(const float* __restrict__ x,
                                              const float4* __restrict__ means,
                                              _Float16* __restrict__ sims) {
    __shared__ float4 sMu[S_];
    const int tid = threadIdx.x;
    const int t = blockIdx.x / (HW_ / 256);
    const int pif = (blockIdx.x % (HW_ / 256)) * 256 + tid;   // pixel in frame
    for (int i = tid; i < S_; i += 256) sMu[i] = means[t * S_ + i];
    __syncthreads();

    const int hw = pif;
    const int h = hw / W_;
    const int w = hw - h * W_;
    float f0 = x[(size_t)(t * C_ + 0) * HW_ + hw];
    float f1 = x[(size_t)(t * C_ + 1) * HW_ + hw];
    float f2 = x[(size_t)(t * C_ + 2) * HW_ + hw];
    float fsq = f0 * f0 + f1 * f1 + f2 * f2;

    // pass 1: max logit
    float mx = -3.0e38f;
    for (int s = 0; s < S_; ++s) {
        float4 mu = sMu[s];
        float d2 = fsq + mu.w - 2.0f * (f0 * mu.x + f1 * mu.y + f2 * mu.z);
        d2 = fmaxf(d2, 0.0f);
        mx = fmaxf(mx, -SCALE_ * d2);
    }
    // pass 2: exp, row sum, store unnormalized (values <= 1 -> safe in f16)
    _Float16* hp = sims + ((size_t)(t * HP_ + h + PAD_) * WP_ + (w + PAD_)) * SP_;
    float sum = 0.0f;
    for (int s = 0; s < S_; ++s) {
        float4 mu = sMu[s];
        float d2 = fsq + mu.w - 2.0f * (f0 * mu.x + f1 * mu.y + f2 * mu.z);
        d2 = fmaxf(d2, 0.0f);
        float e = __expf(-SCALE_ * d2 - mx);
        sum += e;
        hp[s] = (_Float16)e;
    }
    for (int s = S_; s < SP_; ++s) hp[s] = (_Float16)0.0f;
    // pass 3: normalize
    float inv = 1.0f / sum;
    for (int s = 0; s < S_; ++s) hp[s] = (_Float16)((float)hp[s] * inv);
}

// ---------------------------------------------------------------------------
// Kernel 5: neighborhood sims-dot-product attention via WMMA.
// One wave handles 16 consecutive pixels along W; per dy it computes two
// 16x16 f32 tiles of A(16xK) x B(Kx16), K = SP_ = 320, as two half-chunks.
// Each half's 20 b128 B-loads are issued as ONE opaque asm clause burst;
// the waitcnt pass tracks the asm defs precisely and inserts partial
// s_wait_loadcnt before each WMMA, so the burst overlaps the 10-deep WMMA
// chain and the next burst. Diagonal band C[m, m+dx] extracted through LDS.
// ---------------------------------------------------------------------------
__global__ __launch_bounds__(256) void k_attn(const _Float16* __restrict__ sims,
                                              float* __restrict__ attn) {
    __shared__ float lds[8][2][16][16];      // per-wave staging of the 2 tiles
    const int tid  = threadIdx.x;
    const int wv   = tid >> 5;
    const int lane = tid & 31;
    const int g    = blockIdx.x * 8 + wv;    // 0 .. 9215 exact
    const int gw   = g % (W_ / 16);
    const int h    = (g / (W_ / 16)) % H_;
    const int t    = g / ((W_ / 16) * H_);
    const int w0   = gw * 16;
    const int g16  = lane >> 4;              // lane half
    const int m    = lane & 15;

    // ---- preload A: rows = 16 pixels (h, w0..w0+15), K = 0..319 ----
    // 16-bit A layout: lane holds row m; half g holds run [g*16,+16) of each
    // 32-wide K chunk (matches B's contiguous K order).
    const _Float16* arow =
        sims + ((size_t)(t * HP_ + h + PAD_) * WP_ + (w0 + PAD_ + m)) * SP_;
    v16h A[10];
#pragma unroll
    for (int it = 0; it < 10; ++it) {
        const int kb = it * 32;
        v8h lo = *reinterpret_cast<const v8h*>(arow + kb + g16 * 16);
        v8h hi = *reinterpret_cast<const v8h*>(arow + kb + g16 * 16 + 8);
        A[it] = CAT16(lo, hi);
    }

    for (int dy = 0; dy < KS_; ++dy) {
        // neighbor row (padded coords): (h + dy - 3) + 3 = h + dy, always valid
        // tile0 columns (padded): w0 + m .. ; tile1 = tile0 + 16 (fits WP_=208)
        const _Float16* brow =
            sims + ((size_t)(t * HP_ + h + dy) * WP_ + (w0 + m)) * SP_;

        // warm the next dy row one step ahead (global_prefetch_b8, near caches)
        if (dy + 1 < KS_) {
            __builtin_prefetch(brow + (size_t)WP_ * SP_, 0, 3);
            __builtin_prefetch(brow + (size_t)(WP_ + 16) * SP_, 0, 3);
        }

        v8f acc0 = zero8f();
        v8f acc1 = zero8f();

#pragma unroll
        for (int half = 0; half < 2; ++half) {
            // per-lane K base: half*160 elements; B layout: lane half = K half
            const _Float16* p0 = brow + half * 160 + g16 * 16;       // tile0
            const _Float16* p1 = p0 + 16 * SP_;                      // tile1
            v8h b0, b1, b2, b3, b4, b5, b6, b7, b8, b9;
            v8h c0, c1, c2, c3, c4, c5, c6, c7, c8, c9;
            // 20-load clause burst, opaque to the scheduler; the waitcnt pass
            // inserts precise partial loadcnt waits at the consumers.
            asm volatile(
                "s_clause 0x13\n\t"
                "global_load_b128 %0,  %20, off\n\t"
                "global_load_b128 %1,  %20, off offset:16\n\t"
                "global_load_b128 %2,  %20, off offset:64\n\t"
                "global_load_b128 %3,  %20, off offset:80\n\t"
                "global_load_b128 %4,  %20, off offset:128\n\t"
                "global_load_b128 %5,  %20, off offset:144\n\t"
                "global_load_b128 %6,  %20, off offset:192\n\t"
                "global_load_b128 %7,  %20, off offset:208\n\t"
                "global_load_b128 %8,  %20, off offset:256\n\t"
                "global_load_b128 %9,  %20, off offset:272\n\t"
                "global_load_b128 %10, %21, off\n\t"
                "global_load_b128 %11, %21, off offset:16\n\t"
                "global_load_b128 %12, %21, off offset:64\n\t"
                "global_load_b128 %13, %21, off offset:80\n\t"
                "global_load_b128 %14, %21, off offset:128\n\t"
                "global_load_b128 %15, %21, off offset:144\n\t"
                "global_load_b128 %16, %21, off offset:192\n\t"
                "global_load_b128 %17, %21, off offset:208\n\t"
                "global_load_b128 %18, %21, off offset:256\n\t"
                "global_load_b128 %19, %21, off offset:272"
                : "=&v"(b0), "=&v"(b1), "=&v"(b2), "=&v"(b3), "=&v"(b4),
                  "=&v"(b5), "=&v"(b6), "=&v"(b7), "=&v"(b8), "=&v"(b9),
                  "=&v"(c0), "=&v"(c1), "=&v"(c2), "=&v"(c3), "=&v"(c4),
                  "=&v"(c5), "=&v"(c6), "=&v"(c7), "=&v"(c8), "=&v"(c9)
                : "v"(p0), "v"(p1));

            v16h B0[5] = { CAT16(b0, b1), CAT16(b2, b3), CAT16(b4, b5),
                           CAT16(b6, b7), CAT16(b8, b9) };
            v16h B1[5] = { CAT16(c0, c1), CAT16(c2, c3), CAT16(c4, c5),
                           CAT16(c6, c7), CAT16(c8, c9) };
            // ---- 10 WMMAs on two independent accumulator chains ----
#pragma unroll
            for (int it = 0; it < 5; ++it) {
                acc0 = __builtin_amdgcn_wmma_f32_16x16x32_f16(
                    false, A[half * 5 + it], false, B0[it], (short)0, acc0, false, false);
                acc1 = __builtin_amdgcn_wmma_f32_16x16x32_f16(
                    false, A[half * 5 + it], false, B1[it], (short)0, acc1, false, false);
            }
        }

        // ---- stage C tiles to LDS (C/D layout: VGPR r -> row r + g16*8) ----
#pragma unroll
        for (int r = 0; r < 8; ++r) {
            lds[wv][0][g16 * 8 + r][m] = acc0[r];
            lds[wv][1][g16 * 8 + r][m] = acc1[r];
        }
        __syncthreads();

        // ---- extract diagonal band: attn[m2, dx] = C[m2, m2+dx] ----
        for (int ii = lane; ii < 16 * KS_; ii += 32) {
            const int m2 = ii / KS_;
            const int dx = ii % KS_;
            const int col = m2 + dx;
            float v = (col < 16) ? lds[wv][0][m2][col] : lds[wv][1][m2][col - 16];
            attn[((size_t)((t * H_ + h) * W_) + w0 + m2) * K2_ + dy * KS_ + dx] = v;
        }
        __syncthreads();
    }
}

// ---------------------------------------------------------------------------
// Kernel 6: max-normalize attention + modulated 7x7 conv epilogue
// ---------------------------------------------------------------------------
__global__ __launch_bounds__(256) void k_out(const float* __restrict__ x,
                                             const float* __restrict__ attn,
                                             const float* __restrict__ Wlin,
                                             const float* __restrict__ blin,
                                             float* __restrict__ out) {
    __shared__ float sW[C_ * C_ * K2_ + C_];     // 441 weights + 3 bias
    const int tid = threadIdx.x;
    for (int i = tid; i < C_ * C_ * K2_ + C_; i += 256)
        sW[i] = (i < C_ * C_ * K2_) ? Wlin[i] : blin[i - C_ * C_ * K2_];
    __syncthreads();

    const int pix = blockIdx.x * 256 + tid;      // 0 .. T*HW-1 exact
    const int t = pix / HW_;
    const int hw = pix - t * HW_;
    const int h = hw / W_;
    const int w = hw - h * W_;

    float a[K2_];
    float mx = -3.0e38f;
    const float* ap = attn + (size_t)pix * K2_;
#pragma unroll
    for (int k = 0; k < K2_; ++k) { a[k] = ap[k]; mx = fmaxf(mx, a[k]); }
    const float inv = 1.0f / (EPS_ + mx);

    float o0 = sW[C_ * C_ * K2_ + 0];
    float o1 = sW[C_ * C_ * K2_ + 1];
    float o2 = sW[C_ * C_ * K2_ + 2];

    int k = 0;
    for (int dy = 0; dy < KS_; ++dy) {
        const int hy = h + dy - PAD_;
        const bool rok = (hy >= 0) && (hy < H_);
#pragma unroll
        for (int dx = 0; dx < KS_; ++dx, ++k) {
            const int wx = w + dx - PAD_;
            if (rok && wx >= 0 && wx < W_) {
                const float rwk = a[k] * inv;
                const size_t base = (size_t)(t * C_ * HW_) + (size_t)hy * W_ + wx;
                const float f0 = x[base];
                const float f1 = x[base + HW_];
                const float f2 = x[base + 2 * HW_];
                o0 += rwk * (sW[0 * 147 + 0 * K2_ + k] * f0 +
                             sW[0 * 147 + 1 * K2_ + k] * f1 +
                             sW[0 * 147 + 2 * K2_ + k] * f2);
                o1 += rwk * (sW[1 * 147 + 0 * K2_ + k] * f0 +
                             sW[1 * 147 + 1 * K2_ + k] * f1 +
                             sW[1 * 147 + 2 * K2_ + k] * f2);
                o2 += rwk * (sW[2 * 147 + 0 * K2_ + k] * f0 +
                             sW[2 * 147 + 1 * K2_ + k] * f1 +
                             sW[2 * 147 + 2 * K2_ + k] * f2);
            }
        }
    }
    out[(size_t)(t * C_ + 0) * HW_ + hw] = o0;
    out[(size_t)(t * C_ + 1) * HW_ + hw] = o1;
    out[(size_t)(t * C_ + 2) * HW_ + hw] = o2;
}

// ---------------------------------------------------------------------------
// Workspace layout (256-byte aligned sections):
//   sums  : T*S*4 f32          @ 0            (19,200 B)
//   means : T*S float4         @ 19,456       (19,200 B)
//   sims  : T*HP*WP*SP f16     @ 38,912       (105,431,040 B)  zero-padded img
//   attn  : T*HW*49 f32        @ 105,469,952  (28,901,376 B)
// total ~134.4 MB
// ---------------------------------------------------------------------------
#define OFF_SUMS 0
#define OFF_MEAN 19456ull
#define OFF_SIMS 38912ull
#define OFF_ATTN 105469952ull
#define SIMS_BYTES ((size_t)T_ * HP_ * WP_ * SP_ * 2)

extern "C" void kernel_launch(void* const* d_in, const int* in_sizes, int n_in,
                              void* d_out, int out_size, void* d_ws, size_t ws_size,
                              hipStream_t stream) {
    const float* x    = (const float*)d_in[0];   // [T,C,H,W]
    const int*   spix = (const int*)  d_in[1];   // [T,H,W]
    const float* Wlin = (const float*)d_in[2];   // [C, C*49]
    const float* blin = (const float*)d_in[3];   // [C]
    // d_in[4] = n_spix (== S_, compile-time constant here)

    char* ws = (char*)d_ws;
    float*     sums  = (float*)    (ws + OFF_SUMS);
    float4*    means = (float4*)   (ws + OFF_MEAN);
    _Float16*  sims  = (_Float16*) (ws + OFF_SIMS);
    float*     attn  = (float*)    (ws + OFF_ATTN);
    float*     out   = (float*)d_out;

    const int nSums = T_ * S_ * 4;
    const size_t n16 = SIMS_BYTES / 16;
    k_fill0<<<4096,                 256, 0, stream>>>((float4*)sims, n16);
    k_zero <<<(nSums + 255) / 256,  256, 0, stream>>>(sums, nSums);
    k_accum<<<(T_ * HW_) / 256,     256, 0, stream>>>(x, spix, sums);
    k_means<<<(T_ * S_ + 255)/256,  256, 0, stream>>>(sums, means);
    k_sims <<<T_ * (HW_ / 256),     256, 0, stream>>>(x, means, sims);
    k_attn <<<(T_ * H_ * (W_/16)) / 8, 256, 0, stream>>>(sims, attn);
    k_out  <<<(T_ * HW_) / 256,     256, 0, stream>>>(x, attn, Wlin, blin, out);
}
